// GCNLinkPredictor_69088843924173
// MI455X (gfx1250) — matrix operand order repaired and checked by reading, not compiled, not run on hardware
//
#include <hip/hip_runtime.h>
#include <hip/hip_bf16.h>
#include <stdint.h>

typedef __attribute__((ext_vector_type(16))) __bf16 v16bf;
typedef __attribute__((ext_vector_type(8)))  float  v8f;

union Frag16 { uint4 u[2]; v16bf v; };

#define HI 512  // HID*IN_CH

// ---------------- K0: init
//  w2t[c][k] = bf16(w2[k][c])            (K-major columns, 512x512)
//  w1t[c][k] = bf16(w1[k][c]), k<16; 0 for k in [16,32)   (K padded to 32)
//  agg = 0 ; deg = 1 (self loop)
__global__ __launch_bounds__(256)
void k_init(const float* __restrict__ w2, const float* __restrict__ w1,
            __bf16* __restrict__ w2t, __bf16* __restrict__ w1t,
            float* __restrict__ agg, float* __restrict__ deg, int N) {
  int i = blockIdx.x * 256 + threadIdx.x;
  if (i < HI * HI) {
    int k = i & (HI - 1);
    int c = i >> 9;
    w2t[i] = (__bf16)w2[k * HI + c];
  }
  if (i < HI * 32) {
    int k = i & 31;
    int c = i >> 5;
    w1t[i] = (k < 16) ? (__bf16)w1[k * HI + c] : (__bf16)0.0f;
  }
  if (i < N * 32) agg[i] = 0.0f;
  if (i < N) deg[i] = 1.0f;
}

// ---------------- K1: fused edge pipeline, 64 edges/block, both GEMMs on WMMA
__global__ __launch_bounds__(256)
void k_edge(const float* __restrict__ x, const long long* __restrict__ ei,
            const float* __restrict__ edge_attr,
            const __bf16* __restrict__ w1t, const float* __restrict__ b1,
            const __bf16* __restrict__ w2t, const float* __restrict__ b2,
            float* __restrict__ agg, float* __restrict__ deg, int E) {
  __shared__ __align__(16) __bf16 h1s[64][520];    // 64 edges x 512 bf16 (+pad)
  __shared__ __align__(16) __bf16 ea_bf[64][40];   // 64 edges x 32 K (16..31 = 0, +pad)
  __shared__ float b1s[512];
  __shared__ float b2s[512];
  __shared__ float x_s[64][16];
  __shared__ float msgs[64][32];
  __shared__ int   srcs[64];
  __shared__ int   dsts[64];

  const int t  = threadIdx.x;
  const int e0 = blockIdx.x * 64;

  if (t < 64) {
    int ee = e0 + t; if (ee >= E) ee = E - 1;
    srcs[t] = (int)ei[ee];
    dsts[t] = (int)ei[(long long)E + ee];
  }
  for (int i = t; i < 512; i += 256) { b1s[i] = b1[i]; b2s[i] = b2[i]; }
  for (int i = t; i < 64 * 32; i += 256) msgs[i >> 5][i & 31] = 0.0f;
  __syncthreads();   // srcs ready
  for (int i = t; i < 64 * 16; i += 256) {
    int e = i >> 4, k = i & 15;
    x_s[e][k] = x[(size_t)srcs[e] * 16 + k];
  }
  for (int i = t; i < 64 * 32; i += 256) {   // ea -> bf16, zero-padded K
    int e = i >> 5, k = i & 31;
    float v = (k < 16 && (e0 + e) < E) ? edge_attr[(size_t)(e0 + e) * 16 + k] : 0.0f;
    ea_bf[e][k] = (__bf16)v;
  }
  __syncthreads();

  const int lane  = t & 31;
  const int wv    = t >> 5;       // 8 waves; wave owns 64 columns, all 4 msubs
  const int khalf = lane >> 4;
  const int l15   = lane & 15;

  // ----- Phase A: h1 = relu(ea @ w1 + b1) via WMMA (K padded to 32) -----
  // Compute all 16 fragments first, then post-process, so the scheduler can
  // interleave the add/relu/cvt/store chains and hide WMMA->VALU hazards.
  {
    Frag16 a1[4];
#pragma unroll
    for (int m = 0; m < 4; ++m) {
      a1[m].u[0] = *(const uint4*)&ea_bf[m * 16 + l15][khalf * 8];
      a1[m].u[1] = *(const uint4*)&ea_bf[m * 16 + l15][16 + khalf * 8];
    }
    v8f c1[4][4];
    float bb[4];
#pragma unroll
    for (int ns = 0; ns < 4; ++ns) {
      int col = wv * 64 + ns * 16 + l15;
      const uint4* bp = (const uint4*)(w1t + (size_t)col * 32 + khalf * 8);
      Frag16 b;
      b.u[0] = bp[0];
      b.u[1] = bp[2];
      bb[ns] = b1s[col];
#pragma unroll
      for (int m = 0; m < 4; ++m) {
        v8f z = {};
        c1[ns][m] = __builtin_amdgcn_wmma_f32_16x16x32_bf16(
            false, a1[m].v, false, b.v, (short)0, z, false, false);
      }
    }
#pragma unroll
    for (int ns = 0; ns < 4; ++ns) {
      int col = wv * 64 + ns * 16 + l15;
#pragma unroll
      for (int m = 0; m < 4; ++m) {
#pragma unroll
        for (int r = 0; r < 8; ++r) {
          int e = m * 16 + r + 8 * khalf;         // C layout: M = r + 8*(lane>>4)
          h1s[e][col] = (__bf16)fmaxf(c1[ns][m][r] + bb[ns], 0.0f);
        }
      }
    }
  }
  __syncthreads();

  // ----- Phase B: [64,512] @ [512,512]; each B frag feeds 4 WMMAs -----
  v8f acc[4][4] = {};
  const __bf16* bbase = w2t + ((size_t)(wv * 64 + l15) * 512 + khalf * 8);
  for (int kb = 0; kb < 512; kb += 32) {
    Frag16 a[4];
#pragma unroll
    for (int m = 0; m < 4; ++m) {
      a[m].u[0] = *(const uint4*)&h1s[m * 16 + l15][kb + khalf * 8];
      a[m].u[1] = *(const uint4*)&h1s[m * 16 + l15][kb + 16 + khalf * 8];
    }
#pragma unroll
    for (int ns = 0; ns < 4; ++ns) {
      const uint4* bp = (const uint4*)(bbase + (size_t)ns * 16 * 512 + kb);
      Frag16 b;
      b.u[0] = bp[0];
      b.u[1] = bp[2];
#pragma unroll
      for (int m = 0; m < 4; ++m)
        acc[m][ns] = __builtin_amdgcn_wmma_f32_16x16x32_bf16(
            false, a[m].v, false, b.v, (short)0, acc[m][ns], false, false);
    }
  }

  // ----- Phase C: h += b2; msg[e][o] += x_src[e][i] * h[e][i*32+o] -----
#pragma unroll
  for (int m = 0; m < 4; ++m) {
#pragma unroll
    for (int r = 0; r < 8; ++r) {
      int e = m * 16 + r + 8 * khalf;
      float lo = 0.0f, hi = 0.0f;
#pragma unroll
      for (int ns = 0; ns < 4; ++ns) {
        int c = wv * 64 + ns * 16 + l15;
        float h = acc[m][ns][r] + b2s[c];
        float v = h * x_s[e][wv * 2 + (ns >> 1)];   // i = c>>5
        if (ns & 1) hi += v; else lo += v;          // o = (c>>4 & 1)*16 + l15
      }
      atomicAdd(&msgs[e][l15], lo);
      atomicAdd(&msgs[e][16 + l15], hi);
    }
  }
  __syncthreads();

  // ----- scatter to agg[dst] + degree count -----
#pragma unroll
  for (int rep = 0; rep < 8; ++rep) {
    int idx = rep * 256 + t;
    int ee = idx >> 5, o = idx & 31;
    if (e0 + ee < E)
      unsafeAtomicAdd(&agg[(size_t)dsts[ee] * 32 + o], msgs[ee][o]);
  }
  if (t < 64 && e0 + t < E)
    unsafeAtomicAdd(&deg[dsts[t]], 1.0f);
}

// ---------------- K2: x1 = relu(agg + x@root + nn_bias); dis = rsqrt(deg)
__global__ __launch_bounds__(256)
void k_node1(const float* __restrict__ x, const float* __restrict__ agg,
             const float* __restrict__ root, const float* __restrict__ nnb,
             const float* __restrict__ deg,
             float* __restrict__ x1, float* __restrict__ dis, int N) {
  int n = blockIdx.x * 8 + (threadIdx.x >> 5);
  int o = threadIdx.x & 31;
  if (n >= N) return;
  float a = agg[(size_t)n * 32 + o] + nnb[o];
#pragma unroll
  for (int j = 0; j < 16; ++j) a = fmaf(x[(size_t)n * 16 + j], root[j * 32 + o], a);
  x1[(size_t)n * 32 + o] = fmaxf(a, 0.0f);
  if (o == 0) dis[n] = rsqrtf(deg[n]);
}

// ---------------- K3: xw = x1 @ gcn_w; x2 = self-loop term + gcn_b
__global__ __launch_bounds__(256)
void k_node2(const float* __restrict__ x1, const float* __restrict__ gw,
             const float* __restrict__ gb, const float* __restrict__ dis,
             float* __restrict__ xw, float* __restrict__ x2, int N) {
  int n = blockIdx.x * 8 + (threadIdx.x >> 5);
  int o = threadIdx.x & 31;
  if (n >= N) return;
  float a = 0.0f;
#pragma unroll
  for (int j = 0; j < 32; ++j) a = fmaf(x1[(size_t)n * 32 + j], gw[j * 32 + o], a);
  xw[(size_t)n * 32 + o] = a;
  float d = dis[n];
  x2[(size_t)n * 32 + o] = a * d * d + gb[o];
}

// ---------------- K4: GCN edge scatter: x2[dst] += xw[src] * dis[src]*dis[dst]
__global__ __launch_bounds__(256)
void k_gcn_scatter(const long long* __restrict__ ei, const float* __restrict__ xw,
                   const float* __restrict__ dis, float* __restrict__ x2, int E) {
  int e = blockIdx.x * 8 + (threadIdx.x >> 5);
  int o = threadIdx.x & 31;
  if (e >= E) return;
  int s = (int)ei[e], d = (int)ei[(long long)E + e];
  float nm = dis[s] * dis[d];
  unsafeAtomicAdd(&x2[(size_t)d * 32 + o], xw[(size_t)s * 32 + o] * nm);
}

// ---------------- K5: edge scorer sigmoid([x2[src]|x2[dst]] @ lin_w + lin_b)
__global__ __launch_bounds__(256)
void k_score(const long long* __restrict__ ei, const float* __restrict__ x2,
             const float* __restrict__ lw, const float* __restrict__ lb,
             float* __restrict__ out, int E) {
  int e = blockIdx.x * 256 + threadIdx.x;
  if (e >= E) return;
  int s = (int)ei[e], d = (int)ei[(long long)E + e];
  float a = lb[0];
#pragma unroll
  for (int i = 0; i < 32; ++i) a = fmaf(x2[(size_t)s * 32 + i], lw[i], a);
#pragma unroll
  for (int i = 0; i < 32; ++i) a = fmaf(x2[(size_t)d * 32 + i], lw[32 + i], a);
  out[e] = 1.0f / (1.0f + __expf(-a));
}

extern "C" void kernel_launch(void* const* d_in, const int* in_sizes, int n_in,
                              void* d_out, int out_size, void* d_ws, size_t ws_size,
                              hipStream_t stream) {
  const float*     x    = (const float*)d_in[0];
  const long long* ei   = (const long long*)d_in[1];
  const float*     ea   = (const float*)d_in[2];
  const float*     w1   = (const float*)d_in[3];
  const float*     b1   = (const float*)d_in[4];
  const float*     w2   = (const float*)d_in[5];
  const float*     b2   = (const float*)d_in[6];
  const float*     root = (const float*)d_in[7];
  const float*     nnb  = (const float*)d_in[8];
  const float*     gw   = (const float*)d_in[9];
  const float*     gb   = (const float*)d_in[10];
  const float*     lw   = (const float*)d_in[11];
  const float*     lb   = (const float*)d_in[12];
  float* out = (float*)d_out;

  const int N = in_sizes[0] / 16;
  const int E = in_sizes[2] / 16;

  char* ws = (char*)d_ws;
  size_t off = 0;
  auto take = [&](size_t bytes) {
    char* p = ws + off;
    off = (off + bytes + 255) & ~(size_t)255;
    return p;
  };
  __bf16* w2t = (__bf16*)take((size_t)HI * HI * 2);
  __bf16* w1t = (__bf16*)take((size_t)HI * 32 * 2);
  float*  agg = (float*)take((size_t)N * 32 * 4);
  float*  deg = (float*)take((size_t)N * 4);
  float*  dis = (float*)take((size_t)N * 4);
  float*  x1  = (float*)take((size_t)N * 32 * 4);
  float*  xw  = (float*)take((size_t)N * 32 * 4);
  float*  x2  = (float*)take((size_t)N * 32 * 4);

  int initTot = (N * 32 > HI * HI) ? N * 32 : HI * HI;
  k_init<<<(initTot + 255) / 256, 256, 0, stream>>>(w2, w1, w2t, w1t, agg, deg, N);
  k_edge<<<(E + 63) / 64, 256, 0, stream>>>(x, ei, ea, w1t, b1, w2t, b2, agg, deg, E);
  k_node1<<<(N + 7) / 8, 256, 0, stream>>>(x, agg, root, nnb, deg, x1, dis, N);
  k_node2<<<(N + 7) / 8, 256, 0, stream>>>(x1, gw, gb, dis, xw, x2, N);
  k_gcn_scatter<<<(E + 7) / 8, 256, 0, stream>>>(ei, xw, dis, x2, E);
  k_score<<<(E + 255) / 256, 256, 0, stream>>>(ei, x2, lw, lb, out, E);
}
